// MultiHeadAttention_61125974557114
// MI455X (gfx1250) — matrix operand order repaired and checked by reading, not compiled, not run on hardware
//
#include <hip/hip_runtime.h>

// Problem constants (match reference)
constexpr int CB = 4, CS = 2048, CE = 512, CH = 8, CDK = 64;
constexpr float LN_EPS = 1e-5f;

typedef __attribute__((ext_vector_type(16))) __bf16 v16bf;
typedef __attribute__((ext_vector_type(2)))  __bf16 v2bf;
typedef __attribute__((ext_vector_type(2)))  float  v2f;
typedef __attribute__((ext_vector_type(8)))  float  v8f;

union Frag {
    v16bf v;
    unsigned u[8];
};

#define WMMA_BF16(Af, Bf, Cf) \
    __builtin_amdgcn_wmma_f32_16x16x32_bf16(false, (Af), false, (Bf), (short)0, (Cf), false, false)

// Native fptrunc -> ISel emits v_cvt_pk_bf16_f32 on gfx1250 (verified in round 3 asm).
__device__ __forceinline__ unsigned short f2bf(float f) {
    union { __bf16 h; unsigned short u; } x;
    x.h = (__bf16)f;
    return x.u;
}
__device__ __forceinline__ unsigned f2bf2(float a, float b) {
    v2f f;
    f.x = a; f.y = b;
    v2bf h = __builtin_convertvector(f, v2bf);
    union { v2bf v; unsigned u; } x;
    x.v = h;
    return x.u;
}

__device__ __forceinline__ void load_frag8(Frag& fr, const unsigned short* p0,
                                           const unsigned short* p1) {
    uint4 x0 = *(const uint4*)p0;
    uint4 x1 = *(const uint4*)p1;
    fr.u[0] = x0.x; fr.u[1] = x0.y; fr.u[2] = x0.z; fr.u[3] = x0.w;
    fr.u[4] = x1.x; fr.u[5] = x1.y; fr.u[6] = x1.z; fr.u[7] = x1.w;
}

// ---------------------------------------------------------------------------
// Kernel 1: per-head projections.  q = X*W[h] + b  for X in {Q,K,V}.
// One wave per 64x16 output block (4 row-tiles share one weight B-fragment).
// q,k stored [BH][S][DK] bf16; v stored transposed [BH][DK][S] bf16.
// ---------------------------------------------------------------------------
__global__ void qkv_proj_kernel(const float* __restrict__ Q, const float* __restrict__ K,
                                const float* __restrict__ V,
                                const float* __restrict__ Wq, const float* __restrict__ bq,
                                const float* __restrict__ Wk, const float* __restrict__ bk,
                                const float* __restrict__ Wv, const float* __restrict__ bv,
                                unsigned short* __restrict__ qb, unsigned short* __restrict__ kb,
                                unsigned short* __restrict__ vbT) {
    const int SG = CS / 64;        // 32 groups of 4 s-tiles
    const int TILES_D = CDK / 16;  // 4
    int wid = blockIdx.x * (blockDim.x >> 5) + (threadIdx.x >> 5);
    int t  = wid;
    int dt = t % TILES_D; t /= TILES_D;
    int sg = t % SG;      t /= SG;
    int bh = t % (CB * CH); t /= (CB * CH);
    int mat = t;
    if (mat >= 3) return;                 // wave-uniform

    const float* X;  const float* W;  const float* bias;
    if (mat == 0)      { X = Q; W = Wq; bias = bq; }
    else if (mat == 1) { X = K; W = Wk; bias = bk; }
    else               { X = V; W = Wv; bias = bv; }

    int b = bh / CH, h = bh % CH;
    int lane = threadIdx.x & 31;
    int g = lane >> 4, mn = lane & 15;

    const float* Xrow[4];
#pragma unroll
    for (int si = 0; si < 4; ++si)
        Xrow[si] = X + ((size_t)b * CS + sg * 64 + si * 16 + mn) * CE;
    const float* Wcol = W + (size_t)h * CE * CDK + (dt * 16 + mn);

    v8f acc[4] = {v8f{}, v8f{}, v8f{}, v8f{}};
    for (int kk = 0; kk < CE / 32; ++kk) {
        int e0 = kk * 32;
        Frag Bv;   // weight fragment, shared by the 4 row-tiles
        {
            const float* wp = Wcol + (size_t)(e0 + 16 * g) * CDK;
#pragma unroll
            for (int v = 0; v < 8; ++v) {
                float w0 = wp[(size_t)(2 * v) * CDK];
                float w1 = wp[(size_t)(2 * v + 1) * CDK];
                Bv.u[v] = f2bf2(w0, w1);
            }
        }
#pragma unroll
        for (int si = 0; si < 4; ++si) {
            Frag A;
            const float4* p0 = (const float4*)(Xrow[si] + e0 + 16 * g);
            float4 a0 = p0[0], a1 = p0[1];
            const float4* p1 = (const float4*)(Xrow[si] + e0 + 16 * g + 8);
            float4 a2 = p1[0], a3 = p1[1];
            A.u[0] = f2bf2(a0.x, a0.y); A.u[1] = f2bf2(a0.z, a0.w);
            A.u[2] = f2bf2(a1.x, a1.y); A.u[3] = f2bf2(a1.z, a1.w);
            A.u[4] = f2bf2(a2.x, a2.y); A.u[5] = f2bf2(a2.z, a2.w);
            A.u[6] = f2bf2(a3.x, a3.y); A.u[7] = f2bf2(a3.z, a3.w);
            acc[si] = WMMA_BF16(A.v, Bv.v, acc[si]);
        }
    }

    float bias_n = bias[h * CDK + dt * 16 + mn];
#pragma unroll
    for (int si = 0; si < 4; ++si) {
        int st = sg * 4 + si;
        if (mat < 2) {
            unsigned short* dst = (mat == 0) ? qb : kb;
            size_t base = ((size_t)bh * CS + st * 16 + 8 * g) * CDK + dt * 16 + mn;
#pragma unroll
            for (int r = 0; r < 8; ++r)
                dst[base + (size_t)r * CDK] = f2bf(acc[si][r] + bias_n);
        } else {
            size_t base = ((size_t)bh * CDK + dt * 16 + mn) * CS + st * 16 + 8 * g;
            uint4 pk;
            pk.x = f2bf2(acc[si][0] + bias_n, acc[si][1] + bias_n);
            pk.y = f2bf2(acc[si][2] + bias_n, acc[si][3] + bias_n);
            pk.z = f2bf2(acc[si][4] + bias_n, acc[si][5] + bias_n);
            pk.w = f2bf2(acc[si][6] + bias_n, acc[si][7] + bias_n);
            *(uint4*)(vbT + base) = pk;
        }
    }
}

// ---------------------------------------------------------------------------
// Kernel 2: flash attention, NQ=4 query tiles (64 queries) per wave so each
// K/V fragment feeds 4 WMMAs from registers (4x less K/V traffic).
// S^T = K*Q^T trick: C-layout of exp(S^T) IS the A-layout of P for P*V.
// ---------------------------------------------------------------------------
__global__ void attn_kernel(const unsigned short* __restrict__ qb,
                            const unsigned short* __restrict__ kb,
                            const unsigned short* __restrict__ vbT,
                            unsigned short* __restrict__ z) {
    const int NQ = 4;
    const int QG = CS / (16 * NQ);  // 32 query groups
    int wid = blockIdx.x * (blockDim.x >> 5) + (threadIdx.x >> 5);
    int qg = wid % QG;
    int bh = wid / QG;
    if (bh >= CB * CH) return;           // wave-uniform
    int b = bh / CH, h = bh % CH;
    int lane = threadIdx.x & 31;
    int g = lane >> 4, mn = lane & 15;

    // Q^T B-fragments (lane mn = query column), loop-invariant
    Frag qa[NQ][2];
#pragma unroll
    for (int iq = 0; iq < NQ; ++iq) {
        const unsigned short* qrow =
            qb + ((size_t)bh * CS + qg * 64 + iq * 16 + mn) * CDK;
#pragma unroll
        for (int j = 0; j < 2; ++j)
            load_frag8(qa[iq][j], qrow + j * 32 + 16 * g, qrow + j * 32 + 16 * g + 8);
    }

    float m_run[NQ], l_run[NQ];
#pragma unroll
    for (int iq = 0; iq < NQ; ++iq) { m_run[iq] = -1e30f; l_run[iq] = 0.0f; }
    v8f acc[NQ][4];
#pragma unroll
    for (int iq = 0; iq < NQ; ++iq)
#pragma unroll
        for (int nt = 0; nt < 4; ++nt) acc[iq][nt] = v8f{};

    const unsigned short* kball = kb + (size_t)bh * CS * CDK;
    const unsigned short* vball = vbT + (size_t)bh * CDK * CS;
    const float scale = 0.125f;  // DK^-0.5

    const unsigned short* kpf = kball + (size_t)mn * CDK + 32 * CDK;
    const unsigned short* vpf = vball + (size_t)mn * CS + 32;

    for (int kc = 0; kc < CS / 32; ++kc) {
        int k0 = kc * 32;
        __builtin_prefetch(kpf, 0, 3);
        __builtin_prefetch(vpf, 0, 3);
        kpf += 32 * CDK;
        vpf += 32;

        float p[2][NQ][8];
#pragma unroll
        for (int t = 0; t < 2; ++t) {
            Frag ka[2];
            const unsigned short* krow = kball + (size_t)(k0 + t * 16 + mn) * CDK;
#pragma unroll
            for (int j = 0; j < 2; ++j)
                load_frag8(ka[j], krow + j * 32 + 16 * g, krow + j * 32 + 16 * g + 8);
#pragma unroll
            for (int iq = 0; iq < NQ; ++iq) {
                v8f sc = {};
                sc = WMMA_BF16(ka[0].v, qa[iq][0].v, sc);
                sc = WMMA_BF16(ka[1].v, qa[iq][1].v, sc);
#pragma unroll
                for (int r = 0; r < 8; ++r) p[t][iq][r] = sc[r] * scale;
            }
        }

        Frag pa[NQ];
#pragma unroll
        for (int iq = 0; iq < NQ; ++iq) {
            float tmax = -1e30f;
#pragma unroll
            for (int r = 0; r < 8; ++r)
                tmax = fmaxf(tmax, fmaxf(p[0][iq][r], p[1][iq][r]));
            tmax = fmaxf(tmax, __shfl_xor(tmax, 16, 32));
            float m_new = fmaxf(m_run[iq], tmax);
            float fac = __expf(m_run[iq] - m_new);
            float ls = 0.0f;
#pragma unroll
            for (int r = 0; r < 8; ++r) {
                p[0][iq][r] = __expf(p[0][iq][r] - m_new);
                p[1][iq][r] = __expf(p[1][iq][r] - m_new);
                ls += p[0][iq][r] + p[1][iq][r];
            }
            ls += __shfl_xor(ls, 16, 32);
            l_run[iq] = l_run[iq] * fac + ls;
            m_run[iq] = m_new;
#pragma unroll
            for (int r = 0; r < 8; ++r) {
                float fr = __shfl(fac, 8 * g + r, 32);
                acc[iq][0][r] *= fr; acc[iq][1][r] *= fr;
                acc[iq][2][r] *= fr; acc[iq][3][r] *= fr;
            }
#pragma unroll
            for (int v = 0; v < 4; ++v) pa[iq].u[v] = f2bf2(p[0][iq][2 * v], p[0][iq][2 * v + 1]);
#pragma unroll
            for (int v = 0; v < 4; ++v) pa[iq].u[4 + v] = f2bf2(p[1][iq][2 * v], p[1][iq][2 * v + 1]);
        }

        // ctx += P * V : each V fragment feeds NQ WMMAs
#pragma unroll
        for (int nt = 0; nt < 4; ++nt) {
            Frag vf;
            const unsigned short* vp = vball + (size_t)(nt * 16 + mn) * CS + k0 + 8 * g;
            load_frag8(vf, vp, vp + 16);
#pragma unroll
            for (int iq = 0; iq < NQ; ++iq)
                acc[iq][nt] = WMMA_BF16(pa[iq].v, vf.v, acc[iq][nt]);
        }
    }

    // normalize and store head-concat into z [B][S][E]
#pragma unroll
    for (int iq = 0; iq < NQ; ++iq) {
#pragma unroll
        for (int r = 0; r < 8; ++r) {
            float lr = __shfl(l_run[iq], 8 * g + r, 32);
            float inv = 1.0f / lr;
            size_t row = (size_t)b * CS + qg * 64 + iq * 16 + 8 * g + r;
#pragma unroll
            for (int nt = 0; nt < 4; ++nt)
                z[row * CE + h * CDK + nt * 16 + mn] = f2bf(acc[iq][nt][r] * inv);
        }
    }
}

// ---------------------------------------------------------------------------
// Kernel 3: y = Q + z*Wf + bf.  One wave per 64x16 output block:
// 4 row-tiles share one Wf B-fragment per k-step.
// ---------------------------------------------------------------------------
__global__ void final_proj_kernel(const unsigned short* __restrict__ z,
                                  const float* __restrict__ Wf, const float* __restrict__ bf_,
                                  const float* __restrict__ Qin, float* __restrict__ y) {
    const int NT = CE / 16;              // 32
    const int MG = (CB * CS) / 64;       // 128 row groups
    int wid = blockIdx.x * (blockDim.x >> 5) + (threadIdx.x >> 5);
    int nt = wid % NT;
    int mg = wid / NT;
    if (mg >= MG) return;                // wave-uniform
    int lane = threadIdx.x & 31;
    int g = lane >> 4, mn = lane & 15;

    const unsigned short* zrow[4];
#pragma unroll
    for (int si = 0; si < 4; ++si)
        zrow[si] = z + ((size_t)mg * 64 + si * 16 + mn) * CE;
    const float* wcol = Wf + nt * 16 + mn;

    v8f acc[4] = {v8f{}, v8f{}, v8f{}, v8f{}};
    for (int kk = 0; kk < CE / 32; ++kk) {
        int e0 = kk * 32;
        Frag Bv;
        {
            const float* wp = wcol + (size_t)(e0 + 16 * g) * CE;
#pragma unroll
            for (int v = 0; v < 8; ++v) {
                float w0 = wp[(size_t)(2 * v) * CE];
                float w1 = wp[(size_t)(2 * v + 1) * CE];
                Bv.u[v] = f2bf2(w0, w1);
            }
        }
#pragma unroll
        for (int si = 0; si < 4; ++si) {
            Frag A;
            load_frag8(A, zrow[si] + e0 + 16 * g, zrow[si] + e0 + 16 * g + 8);
            acc[si] = WMMA_BF16(A.v, Bv.v, acc[si]);
        }
    }
    float bias = bf_[nt * 16 + mn];
#pragma unroll
    for (int si = 0; si < 4; ++si) {
        size_t row0 = (size_t)mg * 64 + si * 16 + 8 * g;
#pragma unroll
        for (int r = 0; r < 8; ++r) {
            size_t idx = (row0 + r) * CE + nt * 16 + mn;
            y[idx] = acc[si][r] + bias + Qin[idx];
        }
    }
}

// ---------------------------------------------------------------------------
// Kernel 4: row LayerNorm, one wave per 512-float row
// ---------------------------------------------------------------------------
__global__ void layernorm_kernel(const float* __restrict__ y,
                                 const float* __restrict__ gamma,
                                 const float* __restrict__ beta,
                                 float* __restrict__ out) {
    int wid = blockIdx.x * (blockDim.x >> 5) + (threadIdx.x >> 5);
    if (wid >= CB * CS) return;          // wave-uniform
    int lane = threadIdx.x & 31;
    const float4* row = (const float4*)(y + (size_t)wid * CE);
    float4 xv[4];
    float s = 0.0f, s2 = 0.0f;
#pragma unroll
    for (int i = 0; i < 4; ++i) {
        float4 v = row[lane + 32 * i];
        xv[i] = v;
        s  += v.x + v.y + v.z + v.w;
        s2 += v.x * v.x + v.y * v.y + v.z * v.z + v.w * v.w;
    }
#pragma unroll
    for (int m = 16; m >= 1; m >>= 1) {
        s  += __shfl_xor(s,  m, 32);
        s2 += __shfl_xor(s2, m, 32);
    }
    float mean = s * (1.0f / CE);
    float var  = s2 * (1.0f / CE) - mean * mean;
    float inv  = rsqrtf(var + LN_EPS);
    float4* orow = (float4*)(out + (size_t)wid * CE);
    const float4* gp = (const float4*)gamma;
    const float4* bp = (const float4*)beta;
#pragma unroll
    for (int i = 0; i < 4; ++i) {
        float4 gv = gp[lane + 32 * i];
        float4 bv = bp[lane + 32 * i];
        float4 o;
        o.x = (xv[i].x - mean) * inv * gv.x + bv.x;
        o.y = (xv[i].y - mean) * inv * gv.y + bv.y;
        o.z = (xv[i].z - mean) * inv * gv.z + bv.z;
        o.w = (xv[i].w - mean) * inv * gv.w + bv.w;
        orow[lane + 32 * i] = o;
    }
}

// ---------------------------------------------------------------------------
extern "C" void kernel_launch(void* const* d_in, const int* in_sizes, int n_in,
                              void* d_out, int out_size, void* d_ws, size_t ws_size,
                              hipStream_t stream) {
    const float* Q  = (const float*)d_in[0];
    const float* K  = (const float*)d_in[1];
    const float* V  = (const float*)d_in[2];
    const float* Wq = (const float*)d_in[3];
    const float* bq = (const float*)d_in[4];
    const float* Wk = (const float*)d_in[5];
    const float* bk = (const float*)d_in[6];
    const float* Wv = (const float*)d_in[7];
    const float* bv = (const float*)d_in[8];
    const float* Wf = (const float*)d_in[9];
    const float* bf = (const float*)d_in[10];
    const float* gamma = (const float*)d_in[11];
    const float* beta  = (const float*)d_in[12];
    float* out = (float*)d_out;

    // workspace layout (bf16 elements): q | k | vT | z  = 32 MiB total
    unsigned short* ws16 = (unsigned short*)d_ws;
    const size_t PH = (size_t)CB * CH * CS * CDK;          // 4,194,304 elems
    unsigned short* qb  = ws16;
    unsigned short* kb  = ws16 + PH;
    unsigned short* vbT = ws16 + 2 * PH;
    unsigned short* zb  = ws16 + 3 * PH;
    // fp32 y reuses the (dead-after-attention) q|k region: exactly 16 MiB
    float* y = (float*)d_ws;

    dim3 blk(128);  // 4 waves per block
    // 1) 3 * 32 * 32 * 4 = 12288 waves / 4 = 3072 blocks
    qkv_proj_kernel<<<3072, blk, 0, stream>>>(Q, K, V, Wq, bq, Wk, bk, Wv, bv, qb, kb, vbT);
    // 2) 32 * 32 = 1024 waves / 4 = 256 blocks
    attn_kernel<<<256, blk, 0, stream>>>(qb, kb, vbT, zb);
    // 3) 128 * 32 = 4096 waves / 4 = 1024 blocks
    final_proj_kernel<<<1024, blk, 0, stream>>>(zb, Wf, bf, Q, y);
    // 4) 8192 rows / 4 = 2048 blocks
    layernorm_kernel<<<2048, blk, 0, stream>>>(y, gamma, beta, out);
}
